// HyperbolicGraphEncoder_28114855919657
// MI455X (gfx1250) — matrix operand order repaired and checked by reading, not compiled, not run on hardware
//
#include <hip/hip_runtime.h>
#include <math.h>

// ---------------------------------------------------------------------------
// HyperbolicGraphEncoder for MI455X (gfx1250, wave32, WMMA).
// Roofline: 256 MB of adjacency traffic dominates (~11 us @ 23.3 TB/s). All
// GEMMs run through v_wmma_f32_16x16x32_bf16 with on-the-fly fp32->bf16
// conversion during LDS staging (f32 accumulate). The A operand (adjacency)
// is double-buffered in LDS via GLOBAL_LOAD_ASYNC_TO_LDS_B128 + ASYNCcnt so
// the next K-tile's DMA overlaps the current tile's WMMAs. Elementwise
// hyperbolic ops are wave-per-row kernels with shfl_xor reductions.
// ---------------------------------------------------------------------------

typedef __attribute__((ext_vector_type(16))) __bf16          v16bf;
typedef __attribute__((ext_vector_type(8)))  __bf16          v8bf;
typedef __attribute__((ext_vector_type(8)))  float           v8f;
typedef __attribute__((ext_vector_type(8)))  unsigned short  ushort8;

__device__ __forceinline__ unsigned short f2bf(float f) {
  // round-to-nearest-even fp32 -> bf16 (bit pattern)
  unsigned int u = __float_as_uint(f);
  unsigned int r = u + 0x7FFFu + ((u >> 16) & 1u);
  return (unsigned short)(r >> 16);
}

// Async DMA of 16 bytes global -> LDS, tracked by ASYNCcnt.
__device__ __forceinline__ void async_b128(unsigned lds_off, const float* g) {
  asm volatile("global_load_async_to_lds_b128 %0, %1, off"
               :: "v"(lds_off), "v"(g) : "memory");
}
__device__ __forceinline__ void wait_async0() {
  asm volatile("s_wait_asynccnt 0x0" ::: "memory");
}

#define TILE_M 128
#define TILE_N 128
#define TILE_K 32

// C[M x 128] = A[M x K] * B, optional bias[col] and per-row mask.
// B source layout: bTrans==0 -> B stored [N][K] (e.g. W_embed[h][d]);
//                  bTrans==1 -> B stored [K][N] (e.g. W_msg[i][j], m[j][c]).
__global__ __launch_bounds__(256) void gemm_bf16_wmma(
    const float* __restrict__ A, long batchStrideA, int lda,
    const float* __restrict__ B, long batchStrideB, int ldb, int bTrans,
    float* __restrict__ C, long batchStrideC, int ldc,
    int K,
    const float* __restrict__ bias,
    const float* __restrict__ mask, long batchStrideMask)
{
  __shared__ __align__(32) unsigned short ldsA[TILE_M * TILE_K]; // [row][k] bf16
  __shared__ __align__(32) unsigned short ldsB[TILE_N * TILE_K]; // [col][k] bf16
  __shared__ __align__(16) float rawA[2][TILE_M * TILE_K];       // fp32 async dest

  const int tid  = threadIdx.x;
  const int lane = tid & 31;
  const int wv   = tid >> 5;      // 0..7
  const int hf   = lane >> 4;     // 0/1 (lane half)
  const int l16  = lane & 15;
  const int row0 = blockIdx.x * TILE_M;
  const long batch = blockIdx.y;

  const float* Ab = A + batch * batchStrideA;
  const float* Bb = B + batch * batchStrideB;
  float*       Cb = C + batch * batchStrideC;

  // Per-thread A staging mapping: 16 contiguous fp32 per tile. The SAME
  // thread that async-loads a region converts it, so s_wait_asynccnt alone
  // orders the producer/consumer (no extra barrier needed for rawA).
  const int rA = tid >> 1;
  const int hA = tid & 1;
  const int aElem = rA * TILE_K + hA * 16;
  const float* aSrcBase = Ab + (long)(row0 + rA) * lda + hA * 16;

  // Prologue: async-load A tile 0 into rawA[0].
  {
    const unsigned dst = (unsigned)(uintptr_t)(&rawA[0][aElem]);
    #pragma unroll
    for (int j = 0; j < 4; ++j) async_b128(dst + 16u * j, aSrcBase + 4 * j);
  }

  v8f acc[8] = {};
  int buf = 0;

  for (int kk = 0; kk < K; kk += TILE_K) {
    // Wait for this tile's async A data (issued last iteration / prologue).
    wait_async0();

    // ---- convert rawA[buf] -> bf16 ldsA (K-contiguous [row][k])
    {
      const float* src = &rawA[buf][aElem];
      ushort8 t0, t1;
      #pragma unroll
      for (int i = 0; i < 8; ++i) t0[i] = f2bf(src[i]);
      #pragma unroll
      for (int i = 0; i < 8; ++i) t1[i] = f2bf(src[8 + i]);
      *(ushort8*)&ldsA[aElem]     = t0;
      *(ushort8*)&ldsA[aElem + 8] = t1;
    }
    // ---- kick off async DMA of the NEXT A tile (overlaps compute below)
    if (kk + TILE_K < K) {
      const unsigned dst = (unsigned)(uintptr_t)(&rawA[buf ^ 1][aElem]);
      const float* g = aSrcBase + kk + TILE_K;
      #pragma unroll
      for (int j = 0; j < 4; ++j) async_b128(dst + 16u * j, g + 4 * j);
    }

    // ---- stage B tile into ldsB[col][k]
    if (!bTrans) {
      const int c = tid >> 1;
      const int h = tid & 1;
      const float* src = Bb + (long)c * ldb + kk + h * 16;
      ushort8 t0, t1;
      #pragma unroll
      for (int i = 0; i < 8; ++i) t0[i] = f2bf(src[i]);
      #pragma unroll
      for (int i = 0; i < 8; ++i) t1[i] = f2bf(src[8 + i]);
      *(ushort8*)&ldsB[c * TILE_K + h * 16]     = t0;
      *(ushort8*)&ldsB[c * TILE_K + h * 16 + 8] = t1;
      if (kk + TILE_K < K) __builtin_prefetch(src + TILE_K, 0, 1); // global_prefetch_b8
    } else {
      const int k0 = tid >> 3;   // 0..31
      const int p  = tid & 7;    // 0..7 -> 16 cols each
      const float* src = Bb + (long)(kk + k0) * ldb + p * 16;
      #pragma unroll
      for (int i = 0; i < 16; ++i)
        ldsB[(p * 16 + i) * TILE_K + k0] = f2bf(src[i]);
      if (kk + TILE_K < K)
        __builtin_prefetch(Bb + (long)(kk + TILE_K + k0) * ldb + p * 16, 0, 1);
    }
    __syncthreads();

    // ---- compute: wave wv owns rows [row0+16*wv, +16), all 128 cols.
    // 16-bit A fragment: lane half -> K base 0/8; elems 8..15 at K+16.
    const int arow = wv * 16 + l16;
    v8bf alo = *(const v8bf*)&ldsA[arow * TILE_K + hf * 8];
    v8bf ahi = *(const v8bf*)&ldsA[arow * TILE_K + hf * 8 + 16];
    v16bf afrag = __builtin_shufflevector(alo, ahi,
        0,1,2,3,4,5,6,7,8,9,10,11,12,13,14,15);
    // Preload ALL B fragments into distinct registers so the 16 ds_load_b128
    // issue back-to-back instead of a wait before every WMMA.
    v16bf bfrag[8];
    #pragma unroll
    for (int n = 0; n < 8; ++n)
      bfrag[n] = __builtin_shufflevector(
          *(const v8bf*)&ldsB[(n * 16 + l16) * TILE_K + hf * 8],
          *(const v8bf*)&ldsB[(n * 16 + l16) * TILE_K + hf * 8 + 16],
          0,1,2,3,4,5,6,7,8,9,10,11,12,13,14,15);
    #pragma unroll
    for (int n = 0; n < 8; ++n)
      acc[n] = __builtin_amdgcn_wmma_f32_16x16x32_bf16(
          false, afrag, false, bfrag[n], (short)0, acc[n], false, false);
    __syncthreads();
    buf ^= 1;
  }

  // ---- epilogue: D layout: VGPR v -> row = hf*8 + v, col = lane&15
  #pragma unroll
  for (int n = 0; n < 8; ++n) {
    const int col = n * 16 + l16;
    const float bv = bias ? bias[col] : 0.0f;
    #pragma unroll
    for (int v = 0; v < 8; ++v) {
      const int row = row0 + wv * 16 + hf * 8 + v;
      float val = acc[n][v] + bv;
      if (mask) val *= mask[batch * batchStrideMask + row];
      Cb[(long)row * ldc + col] = val;
    }
  }
}

// One wave per 128-float row. mode 0: expmap0+proj; mode 1: logmap0;
// mode 2: relu(in + in2) then expmap0+proj.
__global__ __launch_bounds__(256) void rowwise_hyp(
    const float* __restrict__ in, const float* __restrict__ in2,
    float* __restrict__ out, int mode, int nrows)
{
  const int lane = threadIdx.x & 31;
  const int wv   = threadIdx.x >> 5;
  const int row  = blockIdx.x * 8 + wv;
  if (row >= nrows) return;

  float4 v = ((const float4*)(in + (long)row * 128))[lane];
  if (mode == 2) {
    float4 g = ((const float4*)(in2 + (long)row * 128))[lane];
    v.x = fmaxf(v.x + g.x, 0.0f);
    v.y = fmaxf(v.y + g.y, 0.0f);
    v.z = fmaxf(v.z + g.z, 0.0f);
    v.w = fmaxf(v.w + g.w, 0.0f);
  }
  float s = v.x * v.x + v.y * v.y + v.z * v.z + v.w * v.w;
  #pragma unroll
  for (int off = 16; off >= 1; off >>= 1) s += __shfl_xor(s, off, 32);

  float n = fmaxf(sqrtf(s), 1e-7f);
  float scale;
  if (mode == 1) {                       // logmap0 (c=1)
    float a = fminf(n, 1.0f - 1e-7f);
    scale = atanhf(a) / n;
  } else {                               // expmap0 then proj
    float tn = tanhf(n);
    scale = tn / n;
    float nn = fmaxf(tn, 1e-7f);
    const float MAX_NORM = 0.99999f;     // (1 - 1e-5)/sqrt(1)
    if (nn > MAX_NORM) scale *= MAX_NORM / nn;
  }
  v.x *= scale; v.y *= scale; v.z *= scale; v.w *= scale;
  ((float4*)(out + (long)row * 128))[lane] = v;
}

extern "C" void kernel_launch(void* const* d_in, const int* in_sizes, int n_in,
                              void* d_out, int out_size, void* d_ws, size_t ws_size,
                              hipStream_t stream) {
  (void)in_sizes; (void)n_in; (void)out_size; (void)ws_size;
  const float* X     = (const float*)d_in[0]; // [32,1024,128]
  const float* Adj   = (const float*)d_in[1]; // [32,1024,1024]
  const float* Mask  = (const float*)d_in[2]; // [32,1024,1]
  const float* Wemb  = (const float*)d_in[3]; // [128,128] (h,d)
  const float* Wmsg  = (const float*)d_in[4]; // [2,128,128]
  const float* Bmsg  = (const float*)d_in[5]; // [2,128]
  const float* Wagg  = (const float*)d_in[6]; // [2,128,128]
  const float* Bagg  = (const float*)d_in[7]; // [2,128]
  const float* Wproj = (const float*)d_in[8]; // [128,128] (2*lat, d)
  const float* Bproj = (const float*)d_in[9]; // [128]
  float* Out = (float*)d_out;                 // [32,1024,128]

  const int  Bn = 32, N = 1024, D = 128, L = 2;
  const long RN = (long)Bn * N;     // 32768 rows
  const long PLANE = RN * D;        // 4,194,304 floats (16 MB)

  float* ws      = (float*)d_ws;
  float* buf_t   = ws;              // h / t / agg2 (reused)
  float* buf_x   = ws + PLANE;
  float* buf_m   = ws + 2 * PLANE;
  float* buf_agg = ws + 3 * PLANE;

  dim3 blk(256);
  dim3 gW((unsigned)(RN / TILE_M), 1);   // 256 tiles, one matrix
  dim3 gBig((unsigned)(N / TILE_M), Bn); // 8 x 32 batched
  dim3 gRow((unsigned)(RN / 8));         // 4096 blocks, 8 rows/block

  // 1) h = X @ Wemb^T   (B direct: Wemb[c][d] is already [col][K])
  gemm_bf16_wmma<<<gW, blk, 0, stream>>>(X, 0L, D, Wemb, 0L, D, 0,
      buf_t, 0L, D, D, nullptr, nullptr, 0L);
  // 2) x = proj(expmap0(h))
  rowwise_hyp<<<gRow, blk, 0, stream>>>(buf_t, nullptr, buf_x, 0, (int)RN);

  for (int l = 0; l < L; ++l) {
    // t = logmap0(x)
    rowwise_hyp<<<gRow, blk, 0, stream>>>(buf_x, nullptr, buf_t, 1, (int)RN);
    // m = t @ Wmsg[l] + bmsg[l]   (B transposed: Wmsg is [K][N])
    gemm_bf16_wmma<<<gW, blk, 0, stream>>>(buf_t, 0L, D,
        Wmsg + (long)l * D * D, 0L, D, 1,
        buf_m, 0L, D, D, Bmsg + (long)l * D, nullptr, 0L);
    // agg = A[b] @ m[b]   (batched; B = m[b] is [K=node][N=feat] -> transposed)
    gemm_bf16_wmma<<<gBig, blk, 0, stream>>>(Adj, (long)N * N, N,
        buf_m, (long)N * D, D, 1,
        buf_agg, (long)N * D, D, N, nullptr, nullptr, 0L);
    // agg2 = agg @ Wagg[l] + bagg[l]
    gemm_bf16_wmma<<<gW, blk, 0, stream>>>(buf_agg, 0L, D,
        Wagg + (long)l * D * D, 0L, D, 1,
        buf_t, 0L, D, D, Bagg + (long)l * D, nullptr, 0L);
    // x = proj(expmap0(relu(m + agg2)))
    rowwise_hyp<<<gRow, blk, 0, stream>>>(buf_m, buf_t, buf_x, 2, (int)RN);
  }

  // tangent = logmap0(x)
  rowwise_hyp<<<gRow, blk, 0, stream>>>(buf_x, nullptr, buf_t, 1, (int)RN);
  // out = (tangent @ Wproj^T + bproj) * mask   (Wproj[c][d] direct)
  gemm_bf16_wmma<<<gW, blk, 0, stream>>>(buf_t, 0L, D, Wproj, 0L, D, 0,
      Out, 0L, D, D, Bproj, Mask, 0L);
}
